// Clip_Codebook_9337258902176
// MI455X (gfx1250) — matrix operand order repaired and checked by reading, not compiled, not run on hardware
//
#include <hip/hip_runtime.h>

// ---------------------------------------------------------------------------
// CLIP-style text transformer + codebook pooling for MI455X (gfx1250, wave32).
// Heavy GEMMs (codebook scores ~6 TFLOP total) run on v_wmma_f32_16x16x32_bf16
// with f32 accumulation. Global->LDS staging uses CDNA5 async-LDS copies
// (ASYNCcnt) with double-buffered LDS tiles when the builtin is available.
// ---------------------------------------------------------------------------

typedef __bf16 bf16;
typedef __attribute__((ext_vector_type(16))) __bf16 bf16x16;
typedef __attribute__((ext_vector_type(8)))  float  f32x8;
typedef __attribute__((ext_vector_type(4)))  int    i32x4;

union FragU { bf16x16 v; unsigned int u[8]; };

// ---------------- async global->LDS support (gfx1250) ----------------
#if defined(__AMDGCN__) && __has_builtin(__builtin_amdgcn_global_load_async_to_lds_b128)
#define HAVE_ASYNC 1
#else
#define HAVE_ASYNC 0
#endif

__device__ __forceinline__ void cp16(bf16* s, const bf16* g) {
#if HAVE_ASYNC
  typedef __attribute__((address_space(1))) i32x4* gp_t;   // global v4i*
  typedef __attribute__((address_space(3))) i32x4* lp_t;   // LDS    v4i*
  __builtin_amdgcn_global_load_async_to_lds_b128((gp_t)(void*)g, (lp_t)(void*)s, 0, 0);
#else
  *(uint4*)s = *(const uint4*)g;
#endif
}

__device__ __forceinline__ void async_wait0() {
#if HAVE_ASYNC
#if __has_builtin(__builtin_amdgcn_s_wait_asynccnt)
  __builtin_amdgcn_s_wait_asynccnt(0);
#else
  asm volatile("s_wait_asynccnt 0x0" ::: "memory");
#endif
#endif
}

// ---------------- problem constants ----------------
#define LNUM 12
#define TT   77
#define DD   512
#define HH   8
#define DHH  64
#define CC   256
#define KK   49408
#define EE   512
#define BB   256
#define BT   19712          // B*T = 256*77 (multiple of 128)

// ---------------- workspace layout (bytes) ----------------
static constexpr size_t OFF_X    = 0;                                  // f32  [BT,512] residual
static constexpr size_t OFF_LN   = OFF_X    + (size_t)BT*512*4;        // bf16 [BT,512] LN/attn-out scratch
static constexpr size_t OFF_BIG  = OFF_LN   + (size_t)BT*512*2;        // bf16 [BT,2048] qkv / mlp scratch
static constexpr size_t OFF_Q2A  = OFF_BIG  + (size_t)BT*2048*2;       // bf16 [BT,256]
static constexpr size_t OFF_Q2C  = OFF_Q2A  + (size_t)BT*256*2;        // bf16 [BT,256]
static constexpr size_t OFF_CMAX = OFF_Q2C  + (size_t)BT*256*2;        // f32  [B,K]
static constexpr size_t OFF_CBB  = OFF_CMAX + (size_t)BB*KK*4;         // bf16 [K,256] codebook
static constexpr size_t OFF_WP   = OFF_CBB  + (size_t)KK*256*2;        // bf16 weight pool
static constexpr size_t WP_ELEMS = 40370176ull;
static constexpr size_t OFF_EOS  = OFF_WP   + WP_ELEMS*2;              // int  [B]
static constexpr size_t OFF_KEEP = OFF_EOS  + 1024;                    // f32  [B,T]
static constexpr size_t OFF_PG   = OFF_KEEP + (size_t)BB*TT*4;         // bf16 [B,512] gathered eos rows
static constexpr size_t WS_NEEDED= OFF_PG   + (size_t)BB*512*2;

// element offsets inside the bf16 weight pool
static constexpr size_t WE_QKV = 0;
static constexpr size_t WE_OUT = 9437184;
static constexpr size_t WE_FC  = 12582912;
static constexpr size_t WE_PROJ= 25165824;
static constexpr size_t WE_F1  = 37748736;
static constexpr size_t WE_F2  = 39321600;
static constexpr size_t WE_TPT = 40108032;

// =====================================================================
// small helpers
// =====================================================================
__global__ __launch_bounds__(256) void cvt_bf16_kernel(const float* __restrict__ X,
                                                       bf16* __restrict__ Y, size_t n) {
  size_t i = (size_t)blockIdx.x * 256 + threadIdx.x;
  if (i < n) Y[i] = (bf16)X[i];
}

__global__ __launch_bounds__(256) void cvt_bf16_T_kernel(const float* __restrict__ X,
                                                         bf16* __restrict__ Y,
                                                         int rows, int cols) {
  size_t i = (size_t)blockIdx.x * 256 + threadIdx.x;
  if (i < (size_t)rows * cols) {
    int r = (int)(i / cols), c = (int)(i % cols);
    Y[(size_t)c * rows + r] = (bf16)X[i];
  }
}

__global__ __launch_bounds__(128) void embed_kernel(const int* __restrict__ text,
                                                    const float* __restrict__ tok,
                                                    const float* __restrict__ pos,
                                                    float* __restrict__ X) {
  int bt = blockIdx.x;
  int t  = bt % TT;
  int id = text[bt];
  for (int d = threadIdx.x; d < DD; d += 128)
    X[(size_t)bt * DD + d] = tok[(size_t)id * DD + d] + pos[(size_t)t * DD + d];
}

__global__ __launch_bounds__(128) void eoskeep_kernel(const int* __restrict__ text,
                                                      int* __restrict__ eos,
                                                      float* __restrict__ keep) {
  int b = blockIdx.x;
  __shared__ int e;
  if (threadIdx.x == 0) {
    int best = -1, bi = 0;
    for (int t = 0; t < TT; ++t) {
      int v = text[b * TT + t];
      if (v > best) { best = v; bi = t; }
    }
    eos[b] = bi; e = bi;
  }
  __syncthreads();
  for (int t = threadIdx.x; t < TT; t += 128)
    keep[b * TT + t] = (text[b * TT + t] != 0 && t != e) ? 1.f : 0.f;
}

// LayerNorm row kernel: input f32 or bf16, output bf16
template <typename TI>
__global__ __launch_bounds__(128) void ln_kernel(const TI* __restrict__ X,
                                                 const float* __restrict__ g,
                                                 const float* __restrict__ be,
                                                 bf16* __restrict__ Y, int Dm) {
  int row = blockIdx.x, tid = threadIdx.x;
  const TI* xr = X + (size_t)row * Dm;
  __shared__ float red[128];
  float s = 0.f;
  for (int d = tid; d < Dm; d += 128) s += (float)xr[d];
  red[tid] = s; __syncthreads();
  for (int o = 64; o > 0; o >>= 1) { if (tid < o) red[tid] += red[tid + o]; __syncthreads(); }
  float mean = red[0] / Dm; __syncthreads();
  float v = 0.f;
  for (int d = tid; d < Dm; d += 128) { float t = (float)xr[d] - mean; v += t * t; }
  red[tid] = v; __syncthreads();
  for (int o = 64; o > 0; o >>= 1) { if (tid < o) red[tid] += red[tid + o]; __syncthreads(); }
  float rstd = rsqrtf(red[0] / Dm + 1e-5f);
  for (int d = tid; d < Dm; d += 128)
    Y[(size_t)row * Dm + d] = (bf16)(((float)xr[d] - mean) * rstd * g[d] + be[d]);
}

// =====================================================================
// Tiled WMMA GEMM:  Y[M,N] = act( A[M,Kd] @ W[N,Kd]^T + bias )
// block = 256 threads (8 waves, wave32), tile 128(M) x 64(N) x 32(K)
// double-buffered LDS, async global->LDS staging (ASYNCcnt) on gfx1250.
// ACT: 0=store bf16, 1=QuickGELU bf16, 2=exact GELU bf16, 3=+= f32 residual, 4=store f32
// =====================================================================
#define ASTR 40
#define BSTR 40
#define A_TILE (128 * ASTR)
#define B_TILE (64 * BSTR)

template <int ACT>
__global__ __launch_bounds__(256) void gemm_bf16(const bf16* __restrict__ A,
                                                 const bf16* __restrict__ W,
                                                 const float* __restrict__ bias,
                                                 bf16* __restrict__ outb,
                                                 float* __restrict__ outf,
                                                 int M, int N, int Kd) {
  __shared__ bf16 As[2 * A_TILE];
  __shared__ bf16 Bs[2 * B_TILE];
  int tid = threadIdx.x;
  int m_base = blockIdx.x * 128, n_base = blockIdx.y * 64;
  int wave = tid >> 5, lane = tid & 31;
  int wm = wave & 3, wn = wave >> 2;       // 4x2 wave grid, each wave 32x32
  int half = lane >> 4, lr = lane & 15;

  auto stage = [&](int buf, int k0) {
    bf16* as = As + buf * A_TILE;
    bf16* bs = Bs + buf * B_TILE;
#if !HAVE_ASYNC
    if (k0 + 32 < Kd)
      __builtin_prefetch(A + (size_t)(m_base + (tid >> 2)) * Kd + k0 + 32 + ((tid & 3) << 3), 0, 3);
#endif
#pragma unroll
    for (int i = 0; i < 2; ++i) {
      int idx = tid + i * 256;
      int r = idx >> 2, c8 = (idx & 3) << 3;
      cp16(as + r * ASTR + c8, A + (size_t)(m_base + r) * Kd + k0 + c8);
    }
    int r = tid >> 2, c8 = (tid & 3) << 3;
    cp16(bs + r * BSTR + c8, W + (size_t)(n_base + r) * Kd + k0 + c8);
  };

  f32x8 acc[2][2] = {};

  stage(0, 0);
  async_wait0();
  __syncthreads();

  int buf = 0;
  for (int k0 = 0; k0 < Kd; k0 += 32) {
    if (k0 + 32 < Kd) stage(buf ^ 1, k0 + 32);   // overlap with compute below
    const bf16* as = As + buf * A_TILE;
    const bf16* bs = Bs + buf * B_TILE;

    bf16x16 af[2], bfg[2];
#pragma unroll
    for (int f = 0; f < 2; ++f) {
      FragU au;
#pragma unroll
      for (int j = 0; j < 8; ++j) {
        int k = 2 * j + ((j >= 4) ? 8 : 0) + 8 * half;        // ISA A 16x32 layout
        au.u[j] = *(const unsigned int*)(as + (wm * 32 + f * 16 + lr) * ASTR + k);
      }
      af[f] = au.v;
      FragU bu;
#pragma unroll
      for (int j = 0; j < 8; ++j) {
        int k = 2 * j + ((j >= 4) ? 8 : 0) + 8 * half;        // ISA B 32x16 layout
        bu.u[j] = *(const unsigned int*)(bs + (wn * 32 + f * 16 + lr) * BSTR + k);
      }
      bfg[f] = bu.v;
    }
#pragma unroll
    for (int fi = 0; fi < 2; ++fi)
#pragma unroll
      for (int fj = 0; fj < 2; ++fj)
        acc[fi][fj] = __builtin_amdgcn_wmma_f32_16x16x32_bf16(
            false, af[fi], false, bfg[fj], (short)0, acc[fi][fj], false, false);

    async_wait0();
    __syncthreads();
    buf ^= 1;
  }

  // epilogue (C/D layout: vgpr j -> row j+8*half, lane%16 -> col)
#pragma unroll
  for (int fi = 0; fi < 2; ++fi)
#pragma unroll
    for (int fj = 0; fj < 2; ++fj) {
      int n = n_base + wn * 32 + fj * 16 + lr;
      float bv = bias ? bias[n] : 0.f;
#pragma unroll
      for (int j = 0; j < 8; ++j) {
        int m = m_base + wm * 32 + fi * 16 + half * 8 + j;
        float v = acc[fi][fj][j] + bv;
        size_t o = (size_t)m * N + n;
        if constexpr (ACT == 0) { outb[o] = (bf16)v; }
        else if constexpr (ACT == 1) { v = v / (1.f + __expf(-1.702f * v)); outb[o] = (bf16)v; }
        else if constexpr (ACT == 2) { v = 0.5f * v * (1.f + erff(v * 0.70710678f)); outb[o] = (bf16)v; }
        else if constexpr (ACT == 3) { outf[o] += v; }
        else { outf[o] = v; }
      }
    }
}

// =====================================================================
// per-(b,h) causal attention, T=77, DH=64 (cheap: ~3 GFLOP/layer)
// =====================================================================
__global__ __launch_bounds__(128) void attn_kernel(const bf16* __restrict__ QKV,
                                                   bf16* __restrict__ O, float scale) {
  int bh = blockIdx.x, b = bh >> 3, h = bh & 7;
  __shared__ bf16 Qs[TT * 64];
  __shared__ bf16 Ks[TT * 64];
  __shared__ bf16 Vs[TT * 64];
  __shared__ float Sc[TT][80];
  int tid = threadIdx.x;
  for (int idx = tid; idx < TT * 8; idx += 128) {
    int t = idx >> 3, c8 = (idx & 7) << 3;
    const bf16* base = QKV + ((size_t)b * TT + t) * 1536 + h * 64;
    cp16(Qs + t * 64 + c8, base + c8);
    cp16(Ks + t * 64 + c8, base + 512 + c8);
    cp16(Vs + t * 64 + c8, base + 1024 + c8);
  }
  async_wait0();
  __syncthreads();
  if (tid < TT) {
    int t = tid;
    float mx = -1e30f;
    for (int j = 0; j <= t; ++j) {
      float s = 0.f;
      for (int d = 0; d < 64; ++d) s += (float)Qs[t * 64 + d] * (float)Ks[j * 64 + d];
      s *= scale;
      Sc[t][j] = s;
      mx = fmaxf(mx, s);
    }
    float sum = 0.f;
    for (int j = 0; j <= t; ++j) { float e = __expf(Sc[t][j] - mx); Sc[t][j] = e; sum += e; }
    float inv = 1.f / sum;
    bf16* orow = O + ((size_t)b * TT + t) * DD + h * 64;
    for (int d = 0; d < 64; ++d) {
      float a = 0.f;
      for (int j = 0; j <= t; ++j) a += Sc[t][j] * (float)Vs[j * 64 + d];
      orow[d] = (bf16)(a * inv);
    }
  }
}

// =====================================================================
// fused codebook scores + max over T (WMMA). Masked/padded rows are zeroed
// in LDS so the 0 value from keep-masking is reproduced exactly.
// grid = (K/128, B). Each of 8 waves owns 16 codebook columns.
// =====================================================================
#define CB_BTILE (128 * BSTR)

__global__ __launch_bounds__(256) void cbmax_kernel(const bf16* __restrict__ Q2,
                                                    const bf16* __restrict__ CB,
                                                    const float* __restrict__ keep,
                                                    float* __restrict__ cmax,
                                                    float scale) {
  __shared__ bf16 Aq[80 * 264];        // 80 x 256 (+8 pad)
  __shared__ bf16 Bs[2 * CB_BTILE];    // double-buffered 128 x 32 (+8 pad)
  int b = blockIdx.y, nblk = blockIdx.x, tid = threadIdx.x;

  // zero the A panel (masked rows must contribute exactly 0 to the max)
  for (int idx = tid; idx < 80 * 32; idx += 256) {
    int r = idx >> 5, c8 = (idx & 31) << 3;
    *(uint4*)(Aq + r * 264 + c8) = make_uint4(0u, 0u, 0u, 0u);
  }
  __syncthreads();
  // async-load only the kept rows over the zeros
  for (int idx = tid; idx < TT * 32; idx += 256) {
    int r = idx >> 5, c8 = (idx & 31) << 3;
    if (keep[b * TT + r] != 0.f)
      cp16(Aq + r * 264 + c8, Q2 + ((size_t)b * TT + r) * CC + c8);
  }

  auto stageB = [&](int buf, int ks) {
    bf16* bs = Bs + buf * CB_BTILE;
#pragma unroll
    for (int i = 0; i < 2; ++i) {
      int idx = tid + i * 256;
      int r = idx >> 2, c8 = (idx & 3) << 3;
      cp16(bs + r * BSTR + c8, CB + ((size_t)(nblk * 128 + r)) * CC + ks * 32 + c8);
    }
  };

  stageB(0, 0);
  async_wait0();
  __syncthreads();

  int wave = tid >> 5, lane = tid & 31, half = lane >> 4, lr = lane & 15;
  int n0 = wave * 16;
  f32x8 acc[5] = {};

  int buf = 0;
  for (int ks = 0; ks < 8; ++ks) {   // contraction over C=256 in steps of 32
    if (ks < 7) stageB(buf ^ 1, ks + 1);
    const bf16* bs = Bs + buf * CB_BTILE;
    FragU bu;
#pragma unroll
    for (int j = 0; j < 8; ++j)
      bu.u[j] = *(const unsigned int*)(bs + (n0 + lr) * BSTR + 2 * j + ((j >= 4) ? 8 : 0) + 8 * half);
    bf16x16 bfr = bu.v;
#pragma unroll
    for (int mt = 0; mt < 5; ++mt) {
      FragU au;
#pragma unroll
      for (int j = 0; j < 8; ++j) {
        int k = 2 * j + ((j >= 4) ? 8 : 0) + 8 * half;
        au.u[j] = *(const unsigned int*)(Aq + (mt * 16 + lr) * 264 + ks * 32 + k);
      }
      acc[mt] = __builtin_amdgcn_wmma_f32_16x16x32_bf16(
          false, au.v, false, bfr, (short)0, acc[mt], false, false);
    }
    async_wait0();
    __syncthreads();
    buf ^= 1;
  }

  float m = 0.f;  // masked rows contribute exactly 0, matching reference
#pragma unroll
  for (int mt = 0; mt < 5; ++mt)
#pragma unroll
    for (int j = 0; j < 8; ++j) m = fmaxf(m, acc[mt][j]);
  m = fmaxf(m, __shfl_xor(m, 16, 32));   // combine row halves (same column)
  if (lane < 16)
    cmax[(size_t)b * KK + nblk * 128 + n0 + lr] = m * scale;
}

// softmax over K per batch row (in place)
__global__ __launch_bounds__(256) void softmaxK_kernel(float* __restrict__ Wm, int Kd) {
  int b = blockIdx.x, tid = threadIdx.x;
  float* row = Wm + (size_t)b * Kd;
  __shared__ float red[256];
  float m = -3.4e38f;
  for (int k = tid; k < Kd; k += 256) m = fmaxf(m, row[k]);
  red[tid] = m; __syncthreads();
  for (int o = 128; o > 0; o >>= 1) { if (tid < o) red[tid] = fmaxf(red[tid], red[tid + o]); __syncthreads(); }
  m = red[0]; __syncthreads();
  float s = 0.f;
  for (int k = tid; k < Kd; k += 256) s += __expf(row[k] - m);
  red[tid] = s; __syncthreads();
  for (int o = 128; o > 0; o >>= 1) { if (tid < o) red[tid] += red[tid + o]; __syncthreads(); }
  float inv = 1.f / red[0];
  for (int k = tid; k < Kd; k += 256) row[k] = __expf(row[k] - m) * inv;
}

// local[b,c] = sum_k w[b,k]*codebook[k,c]  (f32, codebook is L2 resident)
__global__ __launch_bounds__(256) void local_kernel(const float* __restrict__ Wm,
                                                    const float* __restrict__ CB,
                                                    float* __restrict__ out, int Kd) {
  int b = blockIdx.x, c = threadIdx.x;
  const float* wr = Wm + (size_t)b * Kd;
  float acc = 0.f;
  for (int k = 0; k < Kd; ++k) acc += wr[k] * CB[(size_t)k * CC + c];
  out[(size_t)b * CC + c] = acc;
}

__global__ __launch_bounds__(128) void gather_kernel(const bf16* __restrict__ Xln,
                                                     const int* __restrict__ eos,
                                                     bf16* __restrict__ Pg) {
  int b = blockIdx.x, e = eos[b];
  for (int d = threadIdx.x; d < DD; d += 128)
    Pg[(size_t)b * DD + d] = Xln[((size_t)b * TT + e) * DD + d];
}

// =====================================================================
// host driver
// =====================================================================
extern "C" void kernel_launch(void* const* d_in, const int* in_sizes, int n_in,
                              void* d_out, int out_size, void* d_ws, size_t ws_size,
                              hipStream_t stream) {
  (void)in_sizes; (void)n_in; (void)out_size;
  if (ws_size < WS_NEEDED) return;

  const int*   text     = (const int*)  d_in[0];
  const float* tok_emb  = (const float*)d_in[1];
  const float* pos_emb  = (const float*)d_in[2];
  const float* ln1_w    = (const float*)d_in[3];
  const float* ln1_b    = (const float*)d_in[4];
  const float* qkv_w    = (const float*)d_in[5];
  const float* qkv_b    = (const float*)d_in[6];
  const float* out_w    = (const float*)d_in[7];
  const float* out_b    = (const float*)d_in[8];
  const float* ln2_w    = (const float*)d_in[9];
  const float* ln2_b    = (const float*)d_in[10];
  const float* fc_w     = (const float*)d_in[11];
  const float* fc_b     = (const float*)d_in[12];
  const float* proj_w   = (const float*)d_in[13];
  const float* proj_b   = (const float*)d_in[14];
  const float* qm_ln1_w = (const float*)d_in[15];
  const float* qm_ln1_b = (const float*)d_in[16];
  const float* qm_fc1_w = (const float*)d_in[17];
  const float* qm_fc1_b = (const float*)d_in[18];
  const float* qm_ln2_w = (const float*)d_in[19];
  const float* qm_ln2_b = (const float*)d_in[20];
  const float* qm_fc2_w = (const float*)d_in[21];
  const float* qm_fc2_b = (const float*)d_in[22];
  const float* lnf_w    = (const float*)d_in[23];
  const float* lnf_b    = (const float*)d_in[24];
  const float* textproj = (const float*)d_in[25];
  const float* codebook = (const float*)d_in[26];

  char* ws = (char*)d_ws;
  float* xf   = (float*)(ws + OFF_X);
  bf16*  lnb  = (bf16*) (ws + OFF_LN);
  bf16*  bigb = (bf16*) (ws + OFF_BIG);
  bf16*  q2a  = (bf16*) (ws + OFF_Q2A);
  bf16*  q2c  = (bf16*) (ws + OFF_Q2C);
  float* cmax = (float*)(ws + OFF_CMAX);
  bf16*  cbb  = (bf16*) (ws + OFF_CBB);
  bf16*  wp   = (bf16*) (ws + OFF_WP);
  int*   eos  = (int*)  (ws + OFF_EOS);
  float* keep = (float*)(ws + OFF_KEEP);
  bf16*  pg   = (bf16*) (ws + OFF_PG);
  float* outp = (float*)d_out;

  auto cvt = [&](const float* src, bf16* dst, size_t n) {
    cvt_bf16_kernel<<<(unsigned)((n + 255) / 256), 256, 0, stream>>>(src, dst, n);
  };

  // one-pass weight conversions (bandwidth-trivial vs 6 TFLOP of matmul)
  cvt(qkv_w,    wp + WE_QKV, (size_t)LNUM * 1536 * 512);
  cvt(out_w,    wp + WE_OUT, (size_t)LNUM * 512 * 512);
  cvt(fc_w,     wp + WE_FC,  (size_t)LNUM * 2048 * 512);
  cvt(proj_w,   wp + WE_PROJ,(size_t)LNUM * 512 * 2048);
  cvt(qm_fc1_w, wp + WE_F1,  (size_t)LNUM * 256 * 512);
  cvt(qm_fc2_w, wp + WE_F2,  (size_t)LNUM * 256 * 256);
  cvt(codebook, cbb,         (size_t)KK * 256);
  cvt_bf16_T_kernel<<<(unsigned)(((size_t)DD * EE + 255) / 256), 256, 0, stream>>>(
      textproj, wp + WE_TPT, DD, EE);

  embed_kernel<<<BT, 128, 0, stream>>>(text, tok_emb, pos_emb, xf);
  eoskeep_kernel<<<BB, 128, 0, stream>>>(text, eos, keep);

  const float scale_att = 0.125f;    // 1/sqrt(64)
  const float scale_cb  = 0.0625f;   // 1/sqrt(256)

  for (int l = 0; l < LNUM; ++l) {
    // ---- attention ----
    ln_kernel<float><<<BT, 128, 0, stream>>>(xf, ln1_w + l * DD, ln1_b + l * DD, lnb, DD);
    gemm_bf16<0><<<dim3(BT / 128, 1536 / 64), 256, 0, stream>>>(
        lnb, wp + WE_QKV + (size_t)l * 1536 * 512, qkv_b + l * 1536, bigb, nullptr, BT, 1536, 512);
    attn_kernel<<<BB * HH, 128, 0, stream>>>(bigb, lnb, scale_att);
    gemm_bf16<3><<<dim3(BT / 128, 512 / 64), 256, 0, stream>>>(
        lnb, wp + WE_OUT + (size_t)l * 512 * 512, out_b + l * DD, nullptr, xf, BT, 512, 512);
    // ---- MLP ----
    ln_kernel<float><<<BT, 128, 0, stream>>>(xf, ln2_w + l * DD, ln2_b + l * DD, lnb, DD);
    gemm_bf16<1><<<dim3(BT / 128, 2048 / 64), 256, 0, stream>>>(
        lnb, wp + WE_FC + (size_t)l * 2048 * 512, fc_b + l * 2048, bigb, nullptr, BT, 2048, 512);
    gemm_bf16<3><<<dim3(BT / 128, 512 / 64), 256, 0, stream>>>(
        bigb, wp + WE_PROJ + (size_t)l * 512 * 2048, proj_b + l * DD, nullptr, xf, BT, 512, 2048);
    // ---- codebook query map ----
    ln_kernel<float><<<BT, 128, 0, stream>>>(xf, qm_ln1_w + l * DD, qm_ln1_b + l * DD, lnb, DD);
    gemm_bf16<2><<<dim3(BT / 128, 256 / 64), 256, 0, stream>>>(
        lnb, wp + WE_F1 + (size_t)l * 256 * 512, qm_fc1_b + l * CC, q2a, nullptr, BT, 256, 512);
    ln_kernel<bf16><<<BT, 128, 0, stream>>>(q2a, qm_ln2_w + l * CC, qm_ln2_b + l * CC, lnb, CC);
    gemm_bf16<0><<<dim3(BT / 128, 256 / 64), 256, 0, stream>>>(
        lnb, wp + WE_F2 + (size_t)l * 256 * 256, qm_fc2_b + l * CC, q2c, nullptr, BT, 256, 256);
    // ---- codebook attention pooling ----
    cbmax_kernel<<<dim3(KK / 128, BB), 256, 0, stream>>>(q2c, cbb, keep, cmax, scale_cb);
    softmaxK_kernel<<<BB, 256, 0, stream>>>(cmax, KK);
    local_kernel<<<BB, 256, 0, stream>>>(
        cmax, codebook, outp + (size_t)BB * EE + (size_t)l * BB * CC, KK);
  }

  // ---- final LN + eos pooling + text projection ----
  ln_kernel<float><<<BT, 128, 0, stream>>>(xf, lnf_w, lnf_b, lnb, DD);
  gather_kernel<<<BB, 128, 0, stream>>>(lnb, eos, pg);
  gemm_bf16<4><<<dim3(BB / 128, EE / 64), 256, 0, stream>>>(
      pg, wp + WE_TPT, nullptr, nullptr, outp, BB, EE, DD);
}
